// NoisyTopkRouter_70205535420532
// MI455X (gfx1250) — matrix operand order repaired and checked by reading, not compiled, not run on hardware
//
#include <hip/hip_runtime.h>
#include <math.h>

// ---- CDNA5 vector types for WMMA ----
typedef __attribute__((ext_vector_type(16))) __bf16 v16bf;
typedef __attribute__((ext_vector_type(8)))  __bf16 v8bf;
typedef __attribute__((ext_vector_type(4)))  __bf16 v4bf;
typedef __attribute__((ext_vector_type(8)))  float  v8f;
typedef __attribute__((ext_vector_type(4)))  float  v4f;

#define NEXPERT 64            // experts per router matrix
#define NCOLS   128           // route(64) ++ noise(64)
#define KC      64            // K-chunk staged in LDS
#define LDSS    (KC + 8)      // padded LDS stride (bf16 elems) -> conflict-free ds_load_b128
#define WAVES   8             // waves per workgroup (wave32) -> 256 threads
#define TOPK    8
#define BUFELEMS (2 * NCOLS * LDSS)        // bf16 elems per double-buffer half
#define BUFBYTES (BUFELEMS * 2)            // 36864 B

// ============================================================================
// Prep: split W_route ++ W_noise (f32) into hi/lo bf16 planes in global scratch.
// Layout: wbf[flavor][col][k], flavor 0 = hi, 1 = lo.  Runs once per launch (1 MB).
// ============================================================================
__global__ __launch_bounds__(256)
void wsplit_kernel(const float* __restrict__ Wr, const float* __restrict__ Wn,
                   __bf16* __restrict__ wbf, int D)
{
    long idx = ((long)blockIdx.x * 256 + threadIdx.x) * 4;   // f32 elem in [0, 128*D)
    int col = (int)(idx / D);
    int k   = (int)(idx % D);
    if (col >= NCOLS) return;
    const float* src = (col < NEXPERT) ? (Wr + (long)col * D + k)
                                       : (Wn + (long)(col - NEXPERT) * D + k);
    v4f w = *(const v4f*)src;
    v4bf h4, l4;
    #pragma unroll
    for (int c = 0; c < 4; ++c) {
        float f = w[c];
        __bf16 h = (__bf16)f;
        h4[c] = h;
        l4[c] = (__bf16)(f - (float)h);
    }
    *(v4bf*)&wbf[(long)col * D + k]                     = h4;
    *(v4bf*)&wbf[(long)NCOLS * D + (long)col * D + k]   = l4;
}

// ============================================================================
// GEMM: ws[token*128 + col] = x[token,:] . W[col,:]
// Weights: async-DMA'd (GLOBAL_LOAD_ASYNC_TO_LDS_B128, double-buffered) from
// pre-split bf16 planes.  bf16x3 (hi*hi + hi*lo + lo*hi) => ~fp32 accuracy.
// ============================================================================
__global__ __launch_bounds__(WAVES * 32)
void router_gemm_kernel(const float* __restrict__ x,
                        const __bf16* __restrict__ wbf,
                        float* __restrict__ ws, int D)
{
    __shared__ __attribute__((aligned(128))) __bf16 wlds[2 * BUFELEMS];  // double buffer

    const int tid  = threadIdx.x;
    const int lane = tid & 31;
    const int wv   = tid >> 5;
    const int hf   = lane >> 4;       // lane half (0/1)
    const int l15  = lane & 15;
    const long tb  = ((long)blockIdx.x * WAVES + wv) * 16;   // wave's token base

    const unsigned ldsbase = (unsigned)(unsigned long long)(uintptr_t)&wlds[0];

    v8f acc[8];
    v8f vzero = {0.f, 0.f, 0.f, 0.f, 0.f, 0.f, 0.f, 0.f};
    #pragma unroll
    for (int t = 0; t < 8; ++t) acc[t] = vzero;

    const float* xrow = x + (tb + l15) * (long)D;
    const int klo = hf * 8;                       // A-layout lane K offset
    const int nchunks = D / KC;

    // Issue one chunk's async weight DMA: 2048 x 16B units, 8 per thread.
    // LDS side gets the bank-conflict padding (per-lane LDS addressing).
    auto issue_chunk = [&](int kc, int bufsel) {
        const int kbase = kc * KC;
        #pragma unroll
        for (int i = 0; i < 8; ++i) {
            int u   = i * (WAVES * 32) + tid;
            int f   = u >> 10;            // flavor
            int rem = u & 1023;
            int col = rem >> 3;
            int w   = rem & 7;            // 8-bf16 unit within chunk row
            const __bf16* g = wbf + ((long)(f * NCOLS + col) * D + kbase + w * 8);
            unsigned lds = ldsbase + (unsigned)bufsel * BUFBYTES
                         + (unsigned)(((f * NCOLS + col) * LDSS + w * 8) * 2);
            asm volatile("global_load_async_to_lds_b128 %0, %1, off"
                         :: "v"(lds), "v"((unsigned long long)(uintptr_t)g)
                         : "memory");
        }
    };

    issue_chunk(0, 0);

    for (int kc = 0; kc < nchunks; ++kc) {
        if (kc + 1 < nchunks) {
            issue_chunk(kc + 1, (kc + 1) & 1);
            // FIFO completion: waiting <=8 guarantees the chunk-kc batch landed.
            asm volatile("s_wait_asynccnt 8" ::: "memory");
        } else {
            asm volatile("s_wait_asynccnt 0" ::: "memory");
        }
        __syncthreads();                       // chunk kc visible to all waves

        const int kbase  = kc * KC;
        const int bufoff = (kc & 1) * BUFELEMS;

        #pragma unroll
        for (int ks = 0; ks < KC / 32; ++ks) {
            const int kg = kbase + ks * 32 + klo;
            // A operand: 16 f32 -> hi/lo bf16 in ISA 16-bit A layout
            v4f a0 = *(const v4f*)(xrow + kg);
            v4f a1 = *(const v4f*)(xrow + kg + 4);
            v4f a2 = *(const v4f*)(xrow + kg + 16);
            v4f a3 = *(const v4f*)(xrow + kg + 20);
            v16bf ahi, alo;
            #pragma unroll
            for (int i = 0; i < 4; ++i) {
                float f;
                f = a0[i]; { __bf16 h=(__bf16)f; ahi[i]    =h; alo[i]    =(__bf16)(f-(float)h); }
                f = a1[i]; { __bf16 h=(__bf16)f; ahi[4+i]  =h; alo[4+i]  =(__bf16)(f-(float)h); }
                f = a2[i]; { __bf16 h=(__bf16)f; ahi[8+i]  =h; alo[8+i]  =(__bf16)(f-(float)h); }
                f = a3[i]; { __bf16 h=(__bf16)f; ahi[12+i] =h; alo[12+i] =(__bf16)(f-(float)h); }
            }
            const int kkb = ks * 32 + hf * 8;   // B-layout: lane holds K 8h..8h+7 and 16+8h..16+8h+7
            #pragma unroll
            for (int tn = 0; tn < 8; ++tn) {
                const int col = tn * 16 + l15;
                const __bf16* bp = &wlds[bufoff + col * LDSS + kkb];
                const __bf16* bq = bp + NCOLS * LDSS;
                v8bf bh0 = *(const v8bf*)bp;
                v8bf bh1 = *(const v8bf*)(bp + 16);
                v8bf bl0 = *(const v8bf*)bq;
                v8bf bl1 = *(const v8bf*)(bq + 16);
                v16bf bhi, blo;
                #pragma unroll
                for (int i = 0; i < 8; ++i) {
                    bhi[i] = bh0[i]; bhi[8 + i] = bh1[i];
                    blo[i] = bl0[i]; blo[8 + i] = bl1[i];
                }
                acc[tn] = __builtin_amdgcn_wmma_f32_16x16x32_bf16(
                            false, ahi, false, bhi, (short)0, acc[tn], false, false);
                acc[tn] = __builtin_amdgcn_wmma_f32_16x16x32_bf16(
                            false, ahi, false, blo, (short)0, acc[tn], false, false);
                acc[tn] = __builtin_amdgcn_wmma_f32_16x16x32_bf16(
                            false, alo, false, bhi, (short)0, acc[tn], false, false);
            }
        }
        __syncthreads();                       // done reading this buffer
    }

    // ---- write logits tile: C layout = VGPR g -> row (hf*8+g), lane l15 -> col ----
    #pragma unroll
    for (int tn = 0; tn < 8; ++tn) {
        const int col = tn * 16 + l15;
        #pragma unroll
        for (int g = 0; g < 8; ++g) {
            long token = tb + hf * 8 + g;
            ws[token * NCOLS + col] = acc[tn][g];
        }
    }
}

// ============================================================================
// Epilogue: noisy logits, stable top-8, masked softmax, indices.
// ============================================================================
__global__ __launch_bounds__(256)
void router_topk_kernel(const float* __restrict__ ws,
                        const float* __restrict__ eps,
                        float* __restrict__ out,
                        int* __restrict__ idxout, int T)
{
    int t = blockIdx.x * blockDim.x + threadIdx.x;
    if (t >= T) return;
    const float* lg = ws  + (long)t * NCOLS;
    const float* ep = eps + (long)t * NEXPERT;

    float nv[NEXPERT];
    #pragma unroll
    for (int e = 0; e < NEXPERT; ++e) {
        float n  = lg[NEXPERT + e];
        float sp = (n > 30.f) ? n : log1pf(expf(n));   // stable softplus
        nv[e] = lg[e] + ep[e] * sp;
    }

    float tv[TOPK]; int ti[TOPK];
    #pragma unroll
    for (int j = 0; j < TOPK; ++j) { tv[j] = -__builtin_inff(); ti[j] = -1; }
    #pragma unroll
    for (int e = 0; e < NEXPERT; ++e) {
        float v = nv[e];
        if (v > tv[TOPK - 1]) {              // strict > keeps lower-index-first ties
            tv[TOPK - 1] = v; ti[TOPK - 1] = e;
            #pragma unroll
            for (int j = TOPK - 1; j > 0; --j) {
                if (tv[j] > tv[j - 1]) {
                    float tmpv = tv[j]; tv[j] = tv[j - 1]; tv[j - 1] = tmpv;
                    int   tmpi = ti[j]; ti[j] = ti[j - 1]; ti[j - 1] = tmpi;
                }
            }
        }
    }

    float vmax = tv[0];
    float s = 0.f;
    #pragma unroll
    for (int j = 0; j < TOPK; ++j) s += expf(tv[j] - vmax);
    float inv = 1.f / s;

    float* orow = out + (long)t * NEXPERT;
    #pragma unroll
    for (int e = 0; e < NEXPERT; ++e) {
        float o = 0.f;
        #pragma unroll
        for (int j = 0; j < TOPK; ++j)
            if (ti[j] == e) o = expf(nv[e] - vmax) * inv;
        orow[e] = o;
    }
    int* irow = idxout + (long)t * TOPK;
    #pragma unroll
    for (int j = 0; j < TOPK; ++j) irow[j] = ti[j];
}

// ============================================================================
extern "C" void kernel_launch(void* const* d_in, const int* in_sizes, int n_in,
                              void* d_out, int out_size, void* d_ws, size_t ws_size,
                              hipStream_t stream)
{
    const float* x  = (const float*)d_in[0];   // [B,S,D] f32
    const float* Wr = (const float*)d_in[1];   // [E,D]   f32
    const float* Wn = (const float*)d_in[2];   // [E,D]   f32
    const float* ep = (const float*)d_in[3];   // [B,S,E] f32

    const int T = in_sizes[3] / NEXPERT;       // B*S tokens
    const int D = in_sizes[0] / T;             // 2048

    float*  ws  = (float*)d_ws;                         // [T,128] logits
    __bf16* wbf = (__bf16*)(ws + (size_t)T * NCOLS);    // [2][128][D] hi/lo planes
    float*  out = (float*)d_out;                        // [T,64] router_output ...
    int*    idx = (int*)(out + (size_t)T * NEXPERT);    // ... then [T,8] indices int32

    wsplit_kernel<<<dim3((NCOLS * D) / (4 * 256)), 256, 0, stream>>>(Wr, Wn, wbf, D);
    router_gemm_kernel<<<dim3(T / (16 * WAVES)), WAVES * 32, 0, stream>>>(x, wbf, ws, D);
    router_topk_kernel<<<dim3((T + 255) / 256), 256, 0, stream>>>(ws, ep, out, idx, T);
}
